// FWLNet_12816182411669
// MI455X (gfx1250) — matrix operand order repaired and checked by reading, not compiled, not run on hardware
//
#include <hip/hip_runtime.h>

#define NN 1024
#define NEDGE 16384
#define NPOS 8192
#define HD 64
#define EPSV 1e-5f

typedef _Float16 h16;
typedef __attribute__((ext_vector_type(8)))  _Float16 v8h;
typedef __attribute__((ext_vector_type(16))) _Float16 v16h;
typedef __attribute__((ext_vector_type(8)))  float    v8f;

// ---------------- WMMA fragment helpers (CDNA5 16x16x32 f16) ----------------
// A 16x32 layout: lane L holds row M=L&15; elements e<8 -> K=e+8*lh, e>=8 -> K=e+8+8*lh
// B 32x16 layout: lane L holds col N=L&15; elements e<8 -> K=e+8*lh, e>=8 -> K=e+8+8*lh
// C/D: VGPR r -> M=r+8*lh, N=lane&15

__device__ inline v16h frag_merge(v8h lo, v8h hi) {
  return __builtin_shufflevector(lo, hi, 0,1,2,3,4,5,6,7,8,9,10,11,12,13,14,15);
}
__device__ inline v16h load_afrag(const h16* __restrict__ row, int kk, int lh) {
  v8h lo = *(const v8h*)(row + kk + 8*lh);
  v8h hi = *(const v8h*)(row + kk + 16 + 8*lh);
  return frag_merge(lo, hi);
}
__device__ inline v16h load_bfrag(const h16* __restrict__ row, int kk, int lh) {
  const h16* p = row + kk + 8*lh;
  return frag_merge(*(const v8h*)p, *(const v8h*)(p + 16));
}
__device__ inline v8f wmma_f16(v16h a, v16h b, v8f c) {
  return __builtin_amdgcn_wmma_f32_16x16x32_f16(false, a, false, b, (short)0, c, false, false);
}

// ---------------- small front-end kernels ----------------
__global__ __launch_bounds__(256) void k_embed(const int* __restrict__ ids,
                                               const float* __restrict__ emb,
                                               float* __restrict__ x0) {
  int idx = blockIdx.x * 256 + threadIdx.x;           // N*64
  int i = idx >> 6, c = idx & 63;
  x0[idx] = emb[ids[i] * 64 + c];
}

__global__ __launch_bounds__(256) void k_linear64(const float* __restrict__ X,
                                                  const float* __restrict__ W,
                                                  float* __restrict__ H) {
  int idx = blockIdx.x * 256 + threadIdx.x;           // N*64
  int i = idx >> 6, o = idx & 63;
  const float* xr = X + i * 64;
  float acc = 0.f;
#pragma unroll 8
  for (int c = 0; c < 64; ++c) acc += xr[c] * W[c * 64 + o];
  H[idx] = acc;
}

__global__ __launch_bounds__(256) void k_deg_init(float* __restrict__ deg) {
  int i = blockIdx.x * 256 + threadIdx.x;
  if (i < NN) deg[i] = 1.0f;                          // self loop
}
__global__ __launch_bounds__(256) void k_deg_acc(const int* __restrict__ dst,
                                                 float* __restrict__ deg) {
  int e = blockIdx.x * 256 + threadIdx.x;
  if (e < NEDGE) atomicAdd(&deg[dst[e]], 1.0f);
}
__global__ __launch_bounds__(256) void k_conv_init(const float* __restrict__ H,
                                                   const float* __restrict__ deg,
                                                   const float* __restrict__ bias,
                                                   float* __restrict__ CV) {
  int idx = blockIdx.x * 256 + threadIdx.x;           // N*64
  int i = idx >> 6, c = idx & 63;
  CV[idx] = H[idx] / deg[i] + bias[c];                // self loop: norm = 1/deg
}
__global__ __launch_bounds__(256) void k_conv_scatter(const int* __restrict__ src,
                                                      const int* __restrict__ dst,
                                                      const float* __restrict__ deg,
                                                      const float* __restrict__ H,
                                                      float* __restrict__ CV) {
  int idx = blockIdx.x * 256 + threadIdx.x;           // E*64
  int e = idx >> 6, c = idx & 63;
  int s = src[e], d = dst[e];
  float nrm = rsqrtf(deg[s]) * rsqrtf(deg[d]);
  atomicAdd(&CV[d * 64 + c], nrm * H[s * 64 + c]);
}

__global__ __launch_bounds__(256) void k_gnorm_relu(const float* __restrict__ in,
                                                    const float* __restrict__ w,
                                                    const float* __restrict__ b,
                                                    const float* __restrict__ a,
                                                    float* __restrict__ outF) {
  __shared__ float s1[256], s2[256];
  int c = blockIdx.x, t = threadIdx.x;
  float l1 = 0.f, l2 = 0.f;
  for (int i = t; i < NN; i += 256) { float v = in[i * 64 + c]; l1 += v; l2 += v * v; }
  s1[t] = l1; s2[t] = l2; __syncthreads();
  for (int s = 128; s > 0; s >>= 1) {
    if (t < s) { s1[t] += s1[t + s]; s2[t] += s2[t + s]; }
    __syncthreads();
  }
  float m = s1[0] * (1.f / NN);
  float al = a[c];
  float var = s2[0] * (1.f / NN) - (2.f * al - al * al) * m * m;
  float rs = rsqrtf(var + EPSV);
  float wc = w[c], bc = b[c];
  for (int i = t; i < NN; i += 256) {
    float v = in[i * 64 + c];
    float y = wc * (v - al * m) * rs + bc;
    outF[i * 64 + c] = fmaxf(y, 0.f);
  }
}

__global__ __launch_bounds__(256) void k_cvt_f32_f16(const float* __restrict__ in,
                                                     h16* __restrict__ out, int n) {
  int idx = blockIdx.x * 256 + threadIdx.x;
  if (idx < n) out[idx] = (h16)in[idx];
}
__global__ __launch_bounds__(256) void k_zero_h16(h16* __restrict__ p, int n) {
  int idx = blockIdx.x * 256 + threadIdx.x;
  if (idx < n) p[idx] = (h16)0.f;
}
__global__ __launch_bounds__(256) void k_adj_set(const int* __restrict__ src,
                                                 const int* __restrict__ dst,
                                                 h16* __restrict__ adjh) {
  int e = blockIdx.x * 256 + threadIdx.x;
  if (e < NEDGE) adjh[(size_t)src[e] * NN + dst[e]] = (h16)1.0f;
}
__global__ __launch_bounds__(256) void k_m3wt(const float* __restrict__ m3W,
                                              h16* __restrict__ m3Wt) {
  int idx = blockIdx.x * 256 + threadIdx.x;           // 64*160
  if (idx >= 64 * 160) return;
  int k = idx / 160, c = idx % 160;
  m3Wt[idx] = (c < 129) ? (h16)m3W[c * 64 + k] : (h16)0.f;
}

// build all 64 scaled-A planes at once: XFs*[o][i][c] = XFf[i][c] * W[c][o]
__global__ __launch_bounds__(256) void k_scale_all(const float* __restrict__ XFf,
                                                   const float* __restrict__ m1W,
                                                   const float* __restrict__ m2W,
                                                   h16* __restrict__ s1,
                                                   h16* __restrict__ s2) {
  int idx = blockIdx.x * 256 + threadIdx.x;           // 64 * N * 64
  int o = idx >> 16;
  int ic = idx & 65535;
  int c = ic & 63;
  float x = XFf[ic];
  s1[idx] = (h16)(x * m1W[c * 64 + o]);
  s2[idx] = (h16)(x * m2W[c * 64 + o]);
}

// ---------------- batched GEMM1 (K=64): x_o = relu(XFs_o @ XF^T + adj*w64 + b) ----
// blockIdx.z = channel o; block = 128 thr (2x2 waves of 32x32) covering 64x64.
__global__ __launch_bounds__(128) void k_gemm1(const h16* __restrict__ A_all,
                                               const h16* __restrict__ Bc,
                                               h16* __restrict__ out_all,
                                               int transposeStore,
                                               const h16* __restrict__ adjh,
                                               const float* __restrict__ Wbase,
                                               const float* __restrict__ biasbase) {
  int o = blockIdx.z;
  const h16* A = A_all + ((size_t)o << 16);           // N*64 per channel
  h16* out = out_all + ((size_t)o << 20);             // N*N per channel
  int tid = threadIdx.x;
  int lane = tid & 31, wv = tid >> 5;
  int lm = lane & 15, lh = (lane >> 4) & 1;
  int i0 = blockIdx.y * 64 + (wv >> 1) * 32;
  int j0 = blockIdx.x * 64 + (wv & 1) * 32;
  v8f acc[2][2] = {};
#pragma unroll
  for (int kk = 0; kk < 64; kk += 32) {
    v16h a0 = load_afrag(A + (size_t)(i0 + lm) * 64, kk, lh);
    v16h a1 = load_afrag(A + (size_t)(i0 + 16 + lm) * 64, kk, lh);
    v16h b0 = load_bfrag(Bc + (size_t)(j0 + lm) * 64, kk, lh);
    v16h b1 = load_bfrag(Bc + (size_t)(j0 + 16 + lm) * 64, kk, lh);
    acc[0][0] = wmma_f16(a0, b0, acc[0][0]);
    acc[0][1] = wmma_f16(a0, b1, acc[0][1]);
    acc[1][0] = wmma_f16(a1, b0, acc[1][0]);
    acc[1][1] = wmma_f16(a1, b1, acc[1][1]);
  }
  float w64 = Wbase[64 * 64 + o];                     // adjacency-channel weight
  float bias = biasbase[o];
#pragma unroll
  for (int ti = 0; ti < 2; ++ti) {
#pragma unroll
    for (int tj = 0; tj < 2; ++tj) {
      int col = j0 + tj * 16 + lm;
      int rowBase = i0 + ti * 16 + 8 * lh;
      if (!transposeStore) {
#pragma unroll
        for (int r = 0; r < 8; ++r) {
          int row = rowBase + r;
          float v = acc[ti][tj][r] + (float)adjh[(size_t)row * NN + col] * w64 + bias;
          out[(size_t)row * NN + col] = (h16)fmaxf(v, 0.f);
        }
      } else {
        v8h pack;
#pragma unroll
        for (int r = 0; r < 8; ++r) {
          int row = rowBase + r;
          float v = acc[ti][tj][r] + (float)adjh[(size_t)row * NN + col] * w64 + bias;
          pack[r] = (h16)fmaxf(v, 0.f);
        }
        *(v8h*)(out + (size_t)col * NN + rowBase) = pack;  // out[col][row..row+7]
      }
    }
  }
}

// ---------------- batched GEMM2 (K=1024): prod_o = x1_o @ x2_o ------------------
// blockIdx.z = channel o; wave tile 64x64 (16 accumulators), block covers 128x128.
__global__ __launch_bounds__(128) void k_gemm2(const h16* __restrict__ x1all,
                                               const h16* __restrict__ x2Tall,
                                               h16* __restrict__ prodall) {
  int o = blockIdx.z;
  const h16* A = x1all + ((size_t)o << 20);
  const h16* B = x2Tall + ((size_t)o << 20);
  h16* out = prodall + ((size_t)o << 20);
  int tid = threadIdx.x;
  int lane = tid & 31, wv = tid >> 5;
  int lm = lane & 15, lh = (lane >> 4) & 1;
  int i0 = blockIdx.y * 128 + (wv >> 1) * 64;
  int j0 = blockIdx.x * 128 + (wv & 1) * 64;
  v8f acc[4][4] = {};
  for (int kk = 0; kk < NN; kk += 32) {
    v16h af[4], bf[4];
#pragma unroll
    for (int t = 0; t < 4; ++t)
      af[t] = load_afrag(A + (size_t)(i0 + t * 16 + lm) * NN, kk, lh);
#pragma unroll
    for (int t = 0; t < 4; ++t)
      bf[t] = load_bfrag(B + (size_t)(j0 + t * 16 + lm) * NN, kk, lh);
#pragma unroll
    for (int ti = 0; ti < 4; ++ti)
#pragma unroll
      for (int tj = 0; tj < 4; ++tj)
        acc[ti][tj] = wmma_f16(af[ti], bf[tj], acc[ti][tj]);
  }
#pragma unroll
  for (int ti = 0; ti < 4; ++ti) {
#pragma unroll
    for (int tj = 0; tj < 4; ++tj) {
      int col = j0 + tj * 16 + lm;
      int rowBase = i0 + ti * 16 + 8 * lh;
#pragma unroll
      for (int r = 0; r < 8; ++r)
        out[(size_t)(rowBase + r) * NN + col] = (h16)acc[ti][tj][r];
    }
  }
}

// ---------------- stage 3: out3[ij][k] = concat(G, adj, prod)[ij][:] @ m3W + m3b ----
// one wave per 16-row (flattened ij) tile; K padded 129 -> 160; 4 N-tiles of 16.
__global__ __launch_bounds__(128) void k_stage3(const float* __restrict__ XFf,
                                                const h16* __restrict__ adjh,
                                                const h16* __restrict__ prod,
                                                const h16* __restrict__ m3Wt,
                                                const float* __restrict__ m3b,
                                                float* __restrict__ out3) {
  int tid = threadIdx.x;
  int lane = tid & 31, wv = tid >> 5;
  int lm = lane & 15, lh = (lane >> 4) & 1;
  int tile = blockIdx.x * 4 + wv;                     // 0..65535
  size_t ij0 = (size_t)tile * 16;
  int i = (int)(ij0 >> 10);
  int jb = (int)(ij0 & 1023);
  const float* xfi = XFf + i * 64;
  const float* xfj = XFf + (size_t)(jb + lm) * 64;
  size_t ijrow = ij0 + lm;

  v8f acc[4];
#pragma unroll
  for (int nt = 0; nt < 4; ++nt) {
    float bv = m3b[nt * 16 + lm];
#pragma unroll
    for (int r = 0; r < 8; ++r) acc[nt][r] = bv;
  }

  for (int ks = 0; ks < 5; ++ks) {
    int kk = ks * 32;
    union { v16h v; h16 e[16]; } Af;
#pragma unroll
    for (int e = 0; e < 16; ++e) {
      int K = kk + ((e < 8) ? e : e + 8) + 8 * lh;
      float val;
      if (K < 64)       val = xfi[K] * xfj[K];
      else if (K == 64) val = (float)adjh[(size_t)i * NN + jb + lm];
      else if (K < 129) val = (float)prod[((size_t)(K - 65) << 20) + ijrow];
      else              val = 0.f;
      Af.e[e] = (h16)val;
    }
#pragma unroll
    for (int nt = 0; nt < 4; ++nt) {
      v16h b = load_bfrag(m3Wt + (size_t)(nt * 16 + lm) * 160, kk, lh);
      acc[nt] = wmma_f16(Af.v, b, acc[nt]);
    }
  }
#pragma unroll
  for (int nt = 0; nt < 4; ++nt) {
#pragma unroll
    for (int r = 0; r < 8; ++r) {
      size_t row = ij0 + r + 8 * lh;
      out3[row * 64 + nt * 16 + lm] = acc[nt][r];
    }
  }
}

// ---------------- stage 4: per-(j,k) stats over i, then gather P pairs ----------
__global__ __launch_bounds__(256) void k_stats(const float* __restrict__ out3,
                                               const float* __restrict__ alpha,
                                               float* __restrict__ meanb,
                                               float* __restrict__ rstdb) {
  int jk = blockIdx.x * 256 + threadIdx.x;            // N*64
  int k = jk & 63;
  float s1 = 0.f, s2 = 0.f;
  for (int i = 0; i < NN; ++i) {
    float v = out3[(size_t)i * (NN * 64) + jk];
    s1 += v; s2 += v * v;
  }
  float m = s1 * (1.f / NN);
  float a = alpha[k];
  float var = s2 * (1.f / NN) - (2.f * a - a * a) * m * m;
  meanb[jk] = m;
  rstdb[jk] = rsqrtf(var + EPSV);
}

__global__ __launch_bounds__(256) void k_final(const float* __restrict__ out3,
                                               const float* __restrict__ meanb,
                                               const float* __restrict__ rstdb,
                                               const int* __restrict__ pos,
                                               const float* __restrict__ w,
                                               const float* __restrict__ b,
                                               const float* __restrict__ a,
                                               const float* __restrict__ ldW,
                                               const float* __restrict__ ldb,
                                               float* __restrict__ out) {
  int p = blockIdx.x * 256 + threadIdx.x;
  if (p >= NPOS) return;
  int i = pos[2 * p], j = pos[2 * p + 1];
  const float* r1 = out3 + ((size_t)i * NN + j) * 64;
  const float* r2 = out3 + ((size_t)j * NN + i) * 64;
  float acc = 0.f;
#pragma unroll 8
  for (int k = 0; k < 64; ++k) {
    float ak = a[k], wk = w[k], bk = b[k];
    float y1 = wk * (r1[k] - ak * meanb[j * 64 + k]) * rstdb[j * 64 + k] + bk;
    float y2 = wk * (r2[k] - ak * meanb[i * 64 + k]) * rstdb[i * 64 + k] + bk;
    y1 = fmaxf(y1, 0.f);
    y2 = fmaxf(y2, 0.f);
    acc += y1 * y2 * ldW[k];
  }
  out[p] = acc + ldb[0];
}

// ---------------- host ----------------
extern "C" void kernel_launch(void* const* d_in, const int* in_sizes, int n_in,
                              void* d_out, int out_size, void* d_ws, size_t ws_size,
                              hipStream_t stream) {
  const int*   x_ids = (const int*)d_in[0];
  const int*   ei    = (const int*)d_in[1];
  const int*   pos   = (const int*)d_in[2];
  const float* emb   = (const float*)d_in[3];
  const float* gW0   = (const float*)d_in[4];
  const float* gb0   = (const float*)d_in[5];
  const float* gnw0  = (const float*)d_in[6];
  const float* gnb0  = (const float*)d_in[7];
  const float* gna0  = (const float*)d_in[8];
  const float* gW1   = (const float*)d_in[9];
  const float* gb1   = (const float*)d_in[10];
  const float* gnw1  = (const float*)d_in[11];
  const float* gnb1  = (const float*)d_in[12];
  const float* gna1  = (const float*)d_in[13];
  const float* m1W   = (const float*)d_in[14];
  const float* m1b   = (const float*)d_in[15];
  const float* m2W   = (const float*)d_in[16];
  const float* m2b   = (const float*)d_in[17];
  const float* m3W   = (const float*)d_in[18];
  const float* m3b   = (const float*)d_in[19];
  const float* gn3w  = (const float*)d_in[20];
  const float* gn3b  = (const float*)d_in[21];
  const float* gn3a  = (const float*)d_in[22];
  const float* ldW   = (const float*)d_in[23];
  const float* ldbp  = (const float*)d_in[24];
  float* out = (float*)d_out;

  const int* src = ei;
  const int* dst = ei + NEDGE;

  char* ws = (char*)d_ws;
  size_t cur = 0;
  auto alloc = [&](size_t bytes) { size_t o = cur; cur += (bytes + 255) & ~(size_t)255; return o; };
  float* x0    = (float*)(ws + alloc((size_t)NN * 64 * 4));
  float* H     = (float*)(ws + alloc((size_t)NN * 64 * 4));
  float* CV    = (float*)(ws + alloc((size_t)NN * 64 * 4));
  float* XL0   = (float*)(ws + alloc((size_t)NN * 64 * 4));
  float* XFf   = (float*)(ws + alloc((size_t)NN * 64 * 4));
  float* deg   = (float*)(ws + alloc((size_t)NN * 4));
  h16*   XFh   = (h16*)(ws + alloc((size_t)NN * 64 * 2));
  h16*   m3Wt  = (h16*)(ws + alloc((size_t)64 * 160 * 2));
  float* meanb = (float*)(ws + alloc((size_t)NN * 64 * 4));
  float* rstdb = (float*)(ws + alloc((size_t)NN * 64 * 4));
  h16*   adjh  = (h16*)(ws + alloc((size_t)NN * NN * 2));
  h16*   XFs1a = (h16*)(ws + alloc((size_t)64 * NN * 64 * 2));
  h16*   XFs2a = (h16*)(ws + alloc((size_t)64 * NN * 64 * 2));
  h16*   x1all = (h16*)(ws + alloc((size_t)64 * NN * NN * 2));
  h16*   x2Tall= (h16*)(ws + alloc((size_t)64 * NN * NN * 2));
  h16*   prod  = (h16*)(ws + alloc((size_t)64 * NN * NN * 2));
  float* out3  = (float*)(ws + alloc((size_t)NN * NN * 64 * 4));
  (void)ws_size; (void)n_in; (void)in_sizes; (void)out_size;

  const int TPB = 256;
  const int gNC = (NN * 64) / TPB;                    // 256 blocks for N*64 grids

  // ---- graph front-end ----
  k_embed<<<gNC, TPB, 0, stream>>>(x_ids, emb, x0);
  k_deg_init<<<(NN + TPB - 1) / TPB, TPB, 0, stream>>>(deg);
  k_deg_acc<<<(NEDGE + TPB - 1) / TPB, TPB, 0, stream>>>(dst, deg);

  // layer 0
  k_linear64<<<gNC, TPB, 0, stream>>>(x0, gW0, H);
  k_conv_init<<<gNC, TPB, 0, stream>>>(H, deg, gb0, CV);
  k_conv_scatter<<<(NEDGE * 64) / TPB, TPB, 0, stream>>>(src, dst, deg, H, CV);
  k_gnorm_relu<<<64, TPB, 0, stream>>>(CV, gnw0, gnb0, gna0, XL0);
  // layer 1
  k_linear64<<<gNC, TPB, 0, stream>>>(XL0, gW1, H);
  k_conv_init<<<gNC, TPB, 0, stream>>>(H, deg, gb1, CV);
  k_conv_scatter<<<(NEDGE * 64) / TPB, TPB, 0, stream>>>(src, dst, deg, H, CV);
  k_gnorm_relu<<<64, TPB, 0, stream>>>(CV, gnw1, gnb1, gna1, XFf);
  k_cvt_f32_f16<<<gNC, TPB, 0, stream>>>(XFf, XFh, NN * 64);

  // adjacency (f16) and transposed/padded m3W
  k_zero_h16<<<(NN * NN) / TPB, TPB, 0, stream>>>(adjh, NN * NN);
  k_adj_set<<<(NEDGE + TPB - 1) / TPB, TPB, 0, stream>>>(src, dst, adjh);
  k_m3wt<<<(64 * 160 + TPB - 1) / TPB, TPB, 0, stream>>>(m3W, m3Wt);

  // ---- batched channel GEMMs (all 64 channels in one launch each) ----
  k_scale_all<<<(64 * NN * 64) / TPB, TPB, 0, stream>>>(XFf, m1W, m2W, XFs1a, XFs2a);
  dim3 g1(NN / 64, NN / 64, 64);                      // 16x16x64, 128 thr
  k_gemm1<<<g1, 128, 0, stream>>>(XFs1a, XFh, x1all, 0, adjh, m1W, m1b);
  k_gemm1<<<g1, 128, 0, stream>>>(XFs2a, XFh, x2Tall, 1, adjh, m2W, m2b);
  dim3 g2(NN / 128, NN / 128, 64);                    // 8x8x64, 128 thr, 64x64/wave
  k_gemm2<<<g2, 128, 0, stream>>>(x1all, x2Tall, prod);

  // ---- stage 3 big GEMM: (N*N x 129) @ (129 x 64) + bias ----
  k_stage3<<<(NN * NN / 16) / 4, 128, 0, stream>>>(XFf, adjh, prod, m3Wt, m3b, out3);

  // ---- stage 4: graph_norm stats over axis 0, relu+transpose-product+gather ----
  k_stats<<<gNC, TPB, 0, stream>>>(out3, gn3a, meanb, rstdb);
  k_final<<<(NPOS + TPB - 1) / TPB, TPB, 0, stream>>>(out3, meanb, rstdb, pos,
                                                      gn3w, gn3b, gn3a, ldW, ldbp, out);
}
